// GradientTransform_5669356833826
// MI455X (gfx1250) — matrix operand order repaired and checked by reading, not compiled, not run on hardware
//
#include <hip/hip_runtime.h>
#include <stdint.h>

typedef __attribute__((ext_vector_type(16))) __bf16 v16bf;
typedef __attribute__((ext_vector_type(8)))  float  v8f;
typedef __attribute__((ext_vector_type(4)))  __bf16 v4bf;

#define EPSV 1e-7f

// ---------------- column statistics (two pass, deterministic) ----------------
__global__ void colstats_partial(const float* __restrict__ X, float* __restrict__ psum,
                                 float* __restrict__ psq, int D, int rowsPerChunk)
{
    int col = blockIdx.x * blockDim.x + threadIdx.x;
    long r0 = (long)blockIdx.y * rowsPerChunk;
    float s = 0.f, ss = 0.f;
    for (int i = 0; i < rowsPerChunk; ++i) {
        float x = X[(r0 + i) * D + col];
        s += x; ss += x * x;
    }
    psum[blockIdx.y * D + col] = s;
    psq [blockIdx.y * D + col] = ss;
}

__global__ void colstats_final(const float* __restrict__ psum, const float* __restrict__ psq,
                               float* __restrict__ mean, float* __restrict__ inv,
                               int D, int chunks, float n)
{
    int col = blockIdx.x * blockDim.x + threadIdx.x;
    float s = 0.f, ss = 0.f;
    for (int c = 0; c < chunks; ++c) { s += psum[c * D + col]; ss += psq[c * D + col]; }
    float m  = s / n;
    float S  = fmaxf(ss - s * m, 0.f);          // sum of squared deviations
    float sd = sqrtf(S / (n - 1.f));
    mean[col] = m;
    inv[col]  = 1.f / (sd + EPSV);
}

// ---------------- normalize fp32 -> bf16 ----------------
__global__ void normalize_bf16(const float* __restrict__ X, const float* __restrict__ mean,
                               const float* __restrict__ inv, __bf16* __restrict__ Y, int D)
{
    size_t base = ((size_t)blockIdx.x * blockDim.x + threadIdx.x) * 4;
    int col = (int)(base % (size_t)D);
    float4 x = *(const float4*)&X[base];
    v4bf y;
    y.x = (__bf16)((x.x - mean[col + 0]) * inv[col + 0]);
    y.y = (__bf16)((x.y - mean[col + 1]) * inv[col + 1]);
    y.z = (__bf16)((x.z - mean[col + 2]) * inv[col + 2]);
    y.w = (__bf16)((x.w - mean[col + 3]) * inv[col + 3]);
    *(v4bf*)&Y[base] = y;
}

// ---------------- weight transpose + convert: W[K][N] f32 -> Wt[N][K] bf16 ----------------
__global__ __launch_bounds__(256) void wtrans(const float* __restrict__ W, __bf16* __restrict__ Wt,
                                              int K, int N)
{
    __shared__ float tile[64][65];
    int nb = blockIdx.x * 64, kb = blockIdx.y * 64;
    for (int idx = threadIdx.x; idx < 64 * 64; idx += 256) {
        int r = idx >> 6, c = idx & 63;
        tile[r][c] = W[(size_t)(kb + r) * N + nb + c];
    }
    __syncthreads();
    for (int idx = threadIdx.x; idx < 64 * 64; idx += 256) {
        int r = idx >> 6, c = idx & 63;          // r: n-local, c: k-local
        Wt[(size_t)(nb + r) * K + kb + c] = (__bf16)tile[c][r];
    }
}

// ---------------- bf16 WMMA GEMM: C = act(A[M][K] * Bt[N][K]^T + bias) ----------------
#define BM 128
#define BN 256
#define BKK 64
#define LDA 72   // padded row stride (elements): 144B rows -> conflict-free b128 reads
#define LDB 72

// Async cache->LDS DMA (CDNA5): dst = per-lane LDS byte addr, off = per-lane 32-bit
// byte offset, base = uniform SGPR-pair base pointer. Tracked by ASYNCcnt.
#define ASYNC_LD128(ldsaddr, voff, base)                                  \
    asm volatile("global_load_async_to_lds_b128 %0, %1, %2"               \
                 :: "v"(ldsaddr), "v"(voff), "s"(base) : "memory")
#define WAIT_ASYNC0() asm volatile("s_wait_asynccnt 0x0" ::: "memory")

template<bool RELU, typename OutT>
__global__ __launch_bounds__(256)
void gemm_bf16(const __bf16* __restrict__ A, const __bf16* __restrict__ Bt,
               const float* __restrict__ bias, OutT* __restrict__ C,
               int M, int N, int K)
{
    __shared__ __bf16 sA[2][BM * LDA];   // 36 KB
    __shared__ __bf16 sB[2][BN * LDB];   // 72 KB

    const int tid   = threadIdx.x;
    const int lane  = tid & 31;
    const int wave  = tid >> 5;
    const int waveM = wave >> 2;     // 0..1
    const int waveN = wave & 3;      // 0..3
    const int half  = lane >> 4;     // 0/1
    const int l16   = lane & 15;

    const size_t bM = (size_t)blockIdx.y * BM;
    const size_t bN = (size_t)blockIdx.x * BN;

    v8f z = {};
    v8f acc[4][4];
#pragma unroll
    for (int i = 0; i < 4; ++i)
#pragma unroll
        for (int j = 0; j < 4; ++j) acc[i][j] = z;

    const int lr = tid >> 3;          // 0..31 (row within 32-row sweep)
    const int lc = (tid & 7) * 8;     // element column (16B granules)

    auto load_tiles = [&](int k0, int buf) {
#pragma unroll
        for (int i = 0; i < 4; ++i) {
            uint32_t lds = (uint32_t)(uintptr_t)&sA[buf][(i * 32 + lr) * LDA + lc];
            uint32_t off = (uint32_t)(((bM + i * 32 + lr) * (size_t)K + k0 + lc) * sizeof(__bf16));
            ASYNC_LD128(lds, off, A);
        }
#pragma unroll
        for (int i = 0; i < 8; ++i) {
            uint32_t lds = (uint32_t)(uintptr_t)&sB[buf][(i * 32 + lr) * LDB + lc];
            uint32_t off = (uint32_t)(((bN + i * 32 + lr) * (size_t)K + k0 + lc) * sizeof(__bf16));
            ASYNC_LD128(lds, off, Bt);
        }
    };

    auto compute = [&](int buf) {
#pragma unroll
        for (int s = 0; s < 2; ++s) {            // two K=32 steps per 64-wide tile
            v16bf af[4], bfr[4];
            // A 16x32 bf16 frag: lanes 0-15 hold K0-15, lanes 16-31 hold K16-31
            const int koA = s * 32 + half * 16;
#pragma unroll
            for (int mt = 0; mt < 4; ++mt) {
                const __bf16* p = &sA[buf][(waveM * 64 + mt * 16 + l16) * LDA + koA];
                union { uint4 q[2]; v16bf v; } u;
                u.q[0] = *(const uint4*)p;        // K = koA .. koA+7
                u.q[1] = *(const uint4*)(p + 8);  // K = koA+8 .. koA+15
                af[mt] = u.v;
            }
            // B 32x16 bf16 frag (from Wt row-major [N][K]): lanes 0-15 hold K0-15, 16-31 hold K16-31
            const int koB = s * 32 + half * 16;
#pragma unroll
            for (int nt = 0; nt < 4; ++nt) {
                const __bf16* p = &sB[buf][(waveN * 64 + nt * 16 + l16) * LDB + koB];
                union { uint4 q[2]; v16bf v; } u;
                u.q[0] = *(const uint4*)p;        // K = koB .. koB+7
                u.q[1] = *(const uint4*)(p + 8);  // K = koB+8 .. koB+15
                bfr[nt] = u.v;
            }
#pragma unroll
            for (int mt = 0; mt < 4; ++mt)
#pragma unroll
                for (int nt = 0; nt < 4; ++nt)
                    acc[mt][nt] = __builtin_amdgcn_wmma_f32_16x16x32_bf16(
                        false, af[mt], false, bfr[nt], (short)0, acc[mt][nt], false, false);
        }
    };

    const int nsteps = K / BKK;
    load_tiles(0, 0);
    WAIT_ASYNC0();
    __syncthreads();
    for (int step = 0; step < nsteps; ++step) {
        if (step + 1 < nsteps) load_tiles((step + 1) * BKK, (step + 1) & 1);
        compute(step & 1);
        WAIT_ASYNC0();           // async DMA for next tile finished writing LDS
        __syncthreads();
    }

    float bias_r[4];
#pragma unroll
    for (int nt = 0; nt < 4; ++nt)
        bias_r[nt] = bias[bN + waveN * 64 + nt * 16 + l16];

#pragma unroll
    for (int mt = 0; mt < 4; ++mt) {
#pragma unroll
        for (int nt = 0; nt < 4; ++nt) {
            size_t col = bN + waveN * 64 + nt * 16 + l16;
#pragma unroll
            for (int j = 0; j < 8; ++j) {
                size_t row = bM + waveM * 64 + mt * 16 + half * 8 + j;
                float v = acc[mt][nt][j] + bias_r[nt];
                if (RELU) v = fmaxf(v, 0.f);
                if constexpr (sizeof(OutT) == 4) C[row * N + col] = v;
                else                             C[row * N + col] = (OutT)(__bf16)v;
            }
        }
    }
}

// ---------------- host orchestration ----------------
extern "C" void kernel_launch(void* const* d_in, const int* in_sizes, int n_in,
                              void* d_out, int out_size, void* d_ws, size_t ws_size,
                              hipStream_t stream)
{
    (void)in_sizes; (void)n_in; (void)out_size; (void)ws_size;
    const float* u   = (const float*)d_in[0];
    const float* v   = (const float*)d_in[1];
    const float* W1u = (const float*)d_in[2];
    const float* b1u = (const float*)d_in[3];
    const float* W2u = (const float*)d_in[4];
    const float* b2u = (const float*)d_in[5];
    const float* W1v = (const float*)d_in[6];
    const float* b1v = (const float*)d_in[7];
    const float* W2v = (const float*)d_in[8];
    const float* b2v = (const float*)d_in[9];

    const int NR = 8192;            // total rows (8*1024)
    const int XD = 1024, XH = 2048; // u dims
    const int DD = 4096, DH = 8192; // v dims

    char*  ws  = (char*)d_ws;
    size_t off = 0;
    auto alloc = [&](size_t bytes) {
        char* p = ws + off;
        off += (bytes + 255) & ~(size_t)255;
        return p;
    };

    float*  meanU = (float*)alloc((size_t)XD * 4);
    float*  invU  = (float*)alloc((size_t)XD * 4);
    float*  meanV = (float*)alloc((size_t)DD * 4);
    float*  invV  = (float*)alloc((size_t)DD * 4);
    float*  psum  = (float*)alloc((size_t)DD * 64 * 4);
    float*  psq   = (float*)alloc((size_t)DD * 64 * 4);
    __bf16* W1uT  = (__bf16*)alloc((size_t)XD * XH * 2);
    __bf16* W2uT  = (__bf16*)alloc((size_t)XD * XH * 2);
    __bf16* W1vT  = (__bf16*)alloc((size_t)DD * DH * 2);
    __bf16* W2vT  = (__bf16*)alloc((size_t)DD * DH * 2);
    __bf16* u_in  = (__bf16*)alloc((size_t)NR * XD * 2);
    __bf16* h1u   = (__bf16*)alloc((size_t)NR * XH * 2);
    __bf16* v_in  = (__bf16*)alloc((size_t)NR * DD * 2);
    __bf16* h1v   = (__bf16*)alloc((size_t)NR * DH * 2);

    float* out1 = (float*)d_out;
    float* out2 = out1 + (size_t)NR * XD;

    const int CH = 64, RPC = NR / CH;  // 64 row-chunks of 128 rows

    // u standardization
    colstats_partial<<<dim3(XD / 256, CH), 256, 0, stream>>>(u, psum, psq, XD, RPC);
    colstats_final  <<<dim3(XD / 256),     256, 0, stream>>>(psum, psq, meanU, invU, XD, CH, (float)NR);
    normalize_bf16  <<<dim3((NR * XD / 4) / 256), 256, 0, stream>>>(u, meanU, invU, u_in, XD);
    // v standardization
    colstats_partial<<<dim3(DD / 256, CH), 256, 0, stream>>>(v, psum, psq, DD, RPC);
    colstats_final  <<<dim3(DD / 256),     256, 0, stream>>>(psum, psq, meanV, invV, DD, CH, (float)NR);
    normalize_bf16  <<<dim3((NR * DD / 4) / 256), 256, 0, stream>>>(v, meanV, invV, v_in, DD);

    // weights -> bf16, transposed to [N][K]
    wtrans<<<dim3(XH / 64, XD / 64), 256, 0, stream>>>(W1u, W1uT, XD, XH);
    wtrans<<<dim3(XD / 64, XH / 64), 256, 0, stream>>>(W2u, W2uT, XH, XD);
    wtrans<<<dim3(DH / 64, DD / 64), 256, 0, stream>>>(W1v, W1vT, DD, DH);
    wtrans<<<dim3(DD / 64, DH / 64), 256, 0, stream>>>(W2v, W2vT, DH, DD);

    // MLPs
    gemm_bf16<true,  __bf16><<<dim3(XH / BN, NR / BM), 256, 0, stream>>>(u_in, W1uT, b1u, h1u,  NR, XH, XD);
    gemm_bf16<false, float ><<<dim3(XD / BN, NR / BM), 256, 0, stream>>>(h1u,  W2uT, b2u, out1, NR, XD, XH);
    gemm_bf16<true,  __bf16><<<dim3(DH / BN, NR / BM), 256, 0, stream>>>(v_in, W1vT, b1v, h1v,  NR, DH, DD);
    gemm_bf16<false, float ><<<dim3(DD / BN, NR / BM), 256, 0, stream>>>(h1v,  W2vT, b2v, out2, NR, DD, DH);
}
